// MagicPoint_index_45801531244728
// MI455X (gfx1250) — matrix operand order repaired and checked by reading, not compiled, not run on hardware
//
#include <hip/hip_runtime.h>
#include <hip/hip_bf16.h>

typedef __attribute__((ext_vector_type(16))) _Float16 v16h;
typedef __attribute__((ext_vector_type(8)))  _Float16 v8h;
typedef __attribute__((ext_vector_type(8)))  float    v8f;

#define CAP      2048
#define KC       600
#define TOPK_N   300

// ---------------------------------------------------------------------------
// Weight pre-pack for channel-last implicit GEMM.
// K-order: K = (r*3+s)*CIN + ci  (a 32-wide K step is contiguous ci).
// f16 fragments in the wave32 WMMA A-layout (16x32 f16, ISA 7.12.2):
//   lanes 0-15  (M=lane):    V0:K0,1  V1:K2,3  V2:K4,5  V3:K6,7  V4..7:K16..23
//   lanes 16-31 (M=lane-16): V0:K8,9  ...                V4..7:K24..31
// Packed layout: [COUT/16][KS][32 lanes][16 halves] (32B aligned per lane).
// ---------------------------------------------------------------------------
__global__ void pack_w_kernel(const float* __restrict__ w, _Float16* __restrict__ wp,
                              int CIN, int COUT) {
    int KT = CIN * 9;
    int KS = (KT + 31) / 32;
    long total = (long)(COUT / 16) * KS * 32 * 16;
    for (long t = (long)blockIdx.x * blockDim.x + threadIdx.x; t < total;
         t += (long)gridDim.x * blockDim.x) {
        int e    = (int)(t & 15);
        int lane = (int)((t >> 4) & 31);
        long rr  = t >> 9;
        int ks   = (int)(rr % KS);
        int mb   = (int)(rr / KS);
        int hiA  = lane >> 4;
        int v    = e >> 1;
        int lo   = e & 1;
        int K    = (v < 4) ? ((hiA ? 8 : 0) + 2 * v + lo)
                           : ((hiA ? 24 : 16) + 2 * (v - 4) + lo);
        K += ks * 32;
        int M = mb * 16 + (lane & 15);
        _Float16 val = (_Float16)0.f;
        if (K < KT) {
            int rs = K / CIN;          // r*3+s
            int ci = K - rs * CIN;
            val = (_Float16)w[(size_t)M * KT + (size_t)ci * 9 + rs];  // OIHW
        }
        wp[t] = val;
    }
}

// 1x1 head weights: [65][256] -> k-major [256][65] f16
__global__ void pack_wpb_kernel(const float* __restrict__ w, _Float16* __restrict__ wp) {
    int t = blockIdx.x * blockDim.x + threadIdx.x;
    if (t < 65 * 256) {
        int c = t % 65;
        int k = t / 65;
        wp[t] = (_Float16)w[c * 256 + k];
    }
}

__global__ void zero_f32_kernel(float* __restrict__ p, size_t n) {
    for (size_t t = (size_t)blockIdx.x * blockDim.x + threadIdx.x; t < n;
         t += (size_t)gridDim.x * blockDim.x)
        p[t] = 0.f;
}

__global__ void init_cnt_kernel(int* cnt, int n) {
    int t = blockIdx.x * blockDim.x + threadIdx.x;
    if (t < n) cnt[t] = 0;
}

// Zero the 1-px ring of a padded (H+2)x(W+2)xC channel-last activation tensor.
__global__ void zero_border_kernel(_Float16* __restrict__ p, int H, int W, int C, int N) {
    int SW = W + 2;
    long rowBand = (long)SW * C;            // one full padded row
    long colBand = (long)H * C;             // one column strip (rows 1..H)
    long per     = 2 * rowBand + 2 * colBand;
    long total   = (long)N * per;
    long imgSz   = (long)(H + 2) * SW * C;
    for (long t = (long)blockIdx.x * blockDim.x + threadIdx.x; t < total;
         t += (long)gridDim.x * blockDim.x) {
        int  n = (int)(t / per);
        long u = t - (long)n * per;
        long idx;
        if (u < 2 * rowBand) {
            int row  = (u < rowBand) ? 0 : (H + 1);
            long rem = (u < rowBand) ? u : (u - rowBand);
            idx = (long)row * SW * C + rem;
        } else {
            long u2  = u - 2 * rowBand;
            int side = (u2 < colBand) ? 0 : 1;
            long rem = side ? (u2 - colBand) : u2;
            int row  = 1 + (int)(rem / C);
            int c    = (int)(rem % C);
            int col  = side ? (W + 1) : 0;
            idx = ((long)row * SW + col) * C + c;
        }
        p[(size_t)n * imgSz + idx] = (_Float16)0.f;
    }
}

// ---------------------------------------------------------------------------
// Layer 1: CIN=1 direct conv + BN + ReLU (compute-trivial), padded CL out.
// ---------------------------------------------------------------------------
__global__ __launch_bounds__(256) void conv1_kernel(const float* __restrict__ x,
                                                    const float* __restrict__ w,
                                                    const float* __restrict__ sc,
                                                    const float* __restrict__ bi,
                                                    _Float16* __restrict__ out,
                                                    int H, int W, int N) {
    __shared__ float wl[64 * 9];
    __shared__ float sl[64], bl[64];
    for (int t = threadIdx.x; t < 576; t += 256) wl[t] = w[t];
    if (threadIdx.x < 64) {
        sl[threadIdx.x] = sc[threadIdx.x];
        bl[threadIdx.x] = bi[threadIdx.x];
    }
    __syncthreads();
    int SW = W + 2;
    size_t imgSz = (size_t)(H + 2) * SW * 64;
    size_t total = (size_t)N * H * W;
    for (size_t t = (size_t)blockIdx.x * blockDim.x + threadIdx.x; t < total;
         t += (size_t)gridDim.x * blockDim.x) {
        int xc   = (int)(t % W);
        size_t r = t / W;
        int yc   = (int)(r % H);
        int n    = (int)(r / H);
        const float* xp = x + (size_t)n * H * W;
        float tap[9];
#pragma unroll
        for (int q = 0; q < 9; ++q) {
            int dy = q / 3 - 1, dx = q % 3 - 1;
            int yy = yc + dy, xx = xc + dx;
            tap[q] = ((unsigned)yy < (unsigned)H && (unsigned)xx < (unsigned)W)
                         ? xp[(size_t)yy * W + xx] : 0.f;
        }
        _Float16* op = out + (size_t)n * imgSz + ((size_t)(yc + 1) * SW + (xc + 1)) * 64;
        for (int co = 0; co < 64; ++co) {
            float acc = 0.f;
#pragma unroll
            for (int q = 0; q < 9; ++q) acc += tap[q] * wl[co * 9 + q];
            float v = acc * sl[co] + bl[co];
            op[co] = (_Float16)(v > 0.f ? v : 0.f);
        }
    }
}

// ---------------------------------------------------------------------------
// Channel-last implicit-GEMM conv3x3 + BN + ReLU via v_wmma_f32_16x16x32_f16.
// Activations are zero-ring padded (H+2)x(W+2)xC -> the hot loop is fully
// branchless: A is one 32B load and B two 16B loads per lane, each from a
// uniform base + 32-bit voffset, chained into one wmma per K step.
// ---------------------------------------------------------------------------
template <int CIN, int COUT>
__global__ __launch_bounds__(256) void conv3x3_cl_wmma(
    const _Float16* __restrict__ in, const _Float16* __restrict__ wp,
    const float* __restrict__ scale, const float* __restrict__ bias,
    _Float16* __restrict__ out, int H, int W, int N) {
    static_assert(CIN == 64 || CIN == 128, "channel-last path needs CIN in {64,128}");
    constexpr int KT = CIN * 9;
    constexpr int KS = KT / 32;            // exact: KT % 32 == 0
    constexpr int SHC = (CIN == 64) ? 6 : 7;
    constexpr int MB = COUT / 32;

    const int n  = blockIdx.z / MB;
    const int cb = blockIdx.z % MB;
    const int y0 = blockIdx.y;
    const int x0 = blockIdx.x * 64;
    const int SW = W + 2;
    const _Float16* inp = in + (size_t)n * (H + 2) * SW * CIN;

    const int lane   = threadIdx.x & 31;
    const int wave   = threadIdx.x >> 5;
    const int mhalf  = wave >> 2;
    const int ntile  = wave & 3;
    const int hi     = lane >> 4;
    const int col    = lane & 15;
    const int mblock = cb * 2 + mhalf;
    const int nloc   = ntile * 16 + col;
    const int px     = x0 + nloc;

    v8f c = {};
#pragma unroll 6
    for (int ks = 0; ks < KS; ++ks) {
        // A fragment: one 32B aligned vector load from packed weights
        v16h a = *(const v16h*)(wp + (unsigned)(((mblock * KS + ks) * 32 + lane) * 16));
        if (ks + 1 < KS)
            __builtin_prefetch(wp + (unsigned)(((mblock * KS + ks + 1) * 32 + lane) * 16),
                               0, 0);

        // B fragment: two 8-half chunks per lane, K = 8*hi+0..7 and 16+8*hi+0..7
        v8h bh[2];
#pragma unroll
        for (int q = 0; q < 2; ++q) {
            const int kb     = ks * 32 + q * 16 + hi * 8;
            const int rs     = kb >> SHC;        // r*3+s (0..8)
            const int cibase = kb & (CIN - 1);
            const int r_     = rs / 3;
            const int s_     = rs - r_ * 3;
            // padded coords: logical (y0-1+r, px-1+s) -> padded (y0+r, px+s)
            const unsigned boff =
                (unsigned)(((y0 + r_) * SW + (px + s_)) * CIN + cibase);
            bh[q] = *(const v8h*)(inp + boff);
        }
        v16h b = __builtin_shufflevector(bh[0], bh[1], 0, 1, 2, 3, 4, 5, 6, 7, 8, 9, 10, 11, 12, 13, 14, 15);

        c = __builtin_amdgcn_wmma_f32_16x16x32_f16(false, a, false, b, (short)0, c,
                                                   false, false);
    }

    // epilogue: D layout (VGPR j -> M = j + 8*hi, N = col); BN + ReLU, 16B store
    if (px < W) {
        v8h o;
        const int cobase = cb * 32 + mhalf * 16 + hi * 8;
#pragma unroll
        for (int j = 0; j < 8; ++j) {
            float v = c[j] * scale[cobase + j] + bias[cobase + j];
            o[j] = (_Float16)(v > 0.f ? v : 0.f);
        }
        _Float16* op = out + (size_t)n * (H + 2) * SW * COUT;
        *(v8h*)(op + (unsigned)(((y0 + 1) * SW + (px + 1)) * COUT + cobase)) = o;
    }
}

// ---------------------------------------------------------------------------
// 2x2 max pool, f16 channel-last, padded in -> padded out
// ---------------------------------------------------------------------------
__global__ void pool2_kernel(const _Float16* __restrict__ in, _Float16* __restrict__ out,
                             int C, int H, int W, int N) {
    int Ho = H / 2, Wo = W / 2;
    int SWi = W + 2, SWo = Wo + 2;
    size_t imgI = (size_t)(H + 2) * SWi * C;
    size_t imgO = (size_t)(Ho + 2) * SWo * C;
    size_t total = (size_t)N * Ho * Wo * C;
    for (size_t t = (size_t)blockIdx.x * blockDim.x + threadIdx.x; t < total;
         t += (size_t)gridDim.x * blockDim.x) {
        int c    = (int)(t % C);
        size_t r = t / C;
        int wo   = (int)(r % Wo);
        r /= Wo;
        int ho = (int)(r % Ho);
        int n  = (int)(r / Ho);
        const _Float16* p =
            in + (size_t)n * imgI + ((size_t)(1 + 2 * ho) * SWi + (1 + 2 * wo)) * C + c;
        float a = (float)p[0], b = (float)p[C];
        float cc = (float)p[(size_t)SWi * C], d = (float)p[(size_t)SWi * C + C];
        out[(size_t)n * imgO + ((size_t)(ho + 1) * SWo + (wo + 1)) * C + c] =
            (_Float16)fmaxf(fmaxf(a, b), fmaxf(cc, d));
    }
}

// ---------------------------------------------------------------------------
// 1x1 conv (256->65) + BN + softmax(65) + drop dustbin + pixel-shuffle r=8
// act is padded channel-last: 256 contiguous halves per pixel.
// ---------------------------------------------------------------------------
__global__ __launch_bounds__(256) void head_kernel(const _Float16* __restrict__ act,
                                                   const _Float16* __restrict__ wkc,
                                                   const float* __restrict__ spb,
                                                   const float* __restrict__ bpb,
                                                   float* __restrict__ prob, int N) {
    const int Hc = 60, Wc = 80, HW = 480 * 640;
    const int SW = Wc + 2;
    const size_t imgSz = (size_t)(Hc + 2) * SW * 256;
    int gid = blockIdx.x * blockDim.x + threadIdx.x;
    if (gid >= N * Hc * Wc) return;
    int wc = gid % Wc;
    int r  = gid / Wc;
    int hc = r % Hc;
    int n  = r / Hc;
    const _Float16* ap = act + (size_t)n * imgSz + ((size_t)(hc + 1) * SW + (wc + 1)) * 256;

    float acc[65];
    for (int c = 0; c < 65; ++c) acc[c] = 0.f;
    for (int k = 0; k < 256; ++k) {
        float a = (float)ap[k];
        const _Float16* wr = wkc + k * 65;
        for (int c = 0; c < 65; ++c) acc[c] += a * (float)wr[c];
    }
    float mx = -1e30f;
    for (int c = 0; c < 65; ++c) {
        acc[c] = acc[c] * spb[c] + bpb[c];
        mx = fmaxf(mx, acc[c]);
    }
    float den = 0.f;
    for (int c = 0; c < 65; ++c) {
        acc[c] = __expf(acc[c] - mx);
        den += acc[c];
    }
    float inv = 1.f / den;
    for (int c = 0; c < 64; ++c) {
        int gr = c >> 3, gc = c & 7;
        prob[(size_t)n * HW + (size_t)(hc * 8 + gr) * 640 + (wc * 8 + gc)] = acc[c] * inv;
    }
}

// ---------------------------------------------------------------------------
// Candidate gather (score > DET_THRESH) per image
// ---------------------------------------------------------------------------
__global__ void gather_kernel(const float* __restrict__ prob, float* __restrict__ cs,
                              int* __restrict__ cidx, int* __restrict__ cnt, int N) {
    const int HW = 480 * 640;
    size_t total = (size_t)N * HW;
    for (size_t t = (size_t)blockIdx.x * blockDim.x + threadIdx.x; t < total;
         t += (size_t)gridDim.x * blockDim.x) {
        float p = prob[t];
        if (p > 0.001f) {
            int n   = (int)(t / HW);
            int pos = atomicAdd(&cnt[n], 1);
            if (pos < CAP) {
                cs[n * CAP + pos]   = p;
                cidx[n * CAP + pos] = (int)(t - (size_t)n * HW);
            }
        }
    }
}

// ---------------------------------------------------------------------------
// Per-image NMS: bitonic sort (desc) in LDS, greedy IoU suppression, top-300
// ---------------------------------------------------------------------------
__global__ __launch_bounds__(1024) void nms_kernel(const float* __restrict__ cs,
                                                   const int* __restrict__ cidx,
                                                   const int* __restrict__ cnt,
                                                   float* __restrict__ prob_nms,
                                                   float* __restrict__ mask) {
    __shared__ float s[CAP];
    __shared__ int   id[CAP];
    __shared__ int   keep[KC];
    const int HW = 480 * 640, W = 640;
    int n = blockIdx.x;
    int m = cnt[n];
    if (m > CAP) m = CAP;
    for (int t = threadIdx.x; t < CAP; t += 1024) {
        bool v = t < m;
        s[t]  = v ? cs[n * CAP + t] : -1e30f;
        id[t] = v ? cidx[n * CAP + t] : 0;
    }
    __syncthreads();

    for (int k = 2; k <= CAP; k <<= 1) {
        for (int j = k >> 1; j > 0; j >>= 1) {
            for (int i = threadIdx.x; i < CAP; i += 1024) {
                int ixj = i ^ j;
                if (ixj > i) {
                    bool up  = (i & k) == 0;
                    float si = s[i], sj = s[ixj];
                    bool sw  = up ? (si < sj) : (si > sj);
                    if (sw) {
                        s[i] = sj; s[ixj] = si;
                        int tt = id[i]; id[i] = id[ixj]; id[ixj] = tt;
                    }
                }
            }
            __syncthreads();
        }
    }

    for (int t = threadIdx.x; t < KC; t += 1024) keep[t] = (s[t] > 0.001f) ? 1 : 0;
    __syncthreads();

    for (int i = 0; i < KC; ++i) {
        if (keep[i]) {
            float ri = (float)(id[i] / W), ci = (float)(id[i] % W);
            for (int j = threadIdx.x; j < KC; j += 1024) {
                if (j > i && keep[j]) {
                    float rj = (float)(id[j] / W), cj = (float)(id[j] % W);
                    float ih = fmaxf(4.f - fabsf(ri - rj), 0.f) *
                               fmaxf(4.f - fabsf(ci - cj), 0.f);
                    if (ih / (32.f - ih) > 0.1f) keep[j] = 0;
                }
            }
        }
        __syncthreads();
    }
    if (threadIdx.x == 0) {
        int c = 0;
        for (int i = 0; i < KC; ++i)
            if (keep[i] && ++c > TOPK_N) keep[i] = 0;
    }
    __syncthreads();
    for (int j = threadIdx.x; j < KC; j += 1024) {
        if (keep[j]) {
            int ix = id[j];
            prob_nms[(size_t)n * HW + ix] = s[j];
            mask[(size_t)n * HW + ix]     = 1.f;
        }
    }
}

// ---------------------------------------------------------------------------
static inline size_t packElems(int CIN, int COUT) {
    int KT = CIN * 9;
    int KS = (KT + 31) / 32;
    return (size_t)(COUT / 16) * KS * 32 * 16;
}

extern "C" void kernel_launch(void* const* d_in, const int* in_sizes, int n_in,
                              void* d_out, int out_size, void* d_ws, size_t ws_size,
                              hipStream_t stream) {
    const int N = 4;
    const size_t HW  = (size_t)480 * 640;
    const size_t NHW = (size_t)N * HW;

    const float* x = (const float*)d_in[0];
    const float *w[8], *sc[8], *bi[8];
    for (int i = 0; i < 8; ++i) {
        w[i]  = (const float*)d_in[1 + 3 * i];
        sc[i] = (const float*)d_in[2 + 3 * i];
        bi[i] = (const float*)d_in[3 + 3 * i];
    }
    const float* wpa = (const float*)d_in[25];
    const float* spa = (const float*)d_in[26];
    const float* bpa = (const float*)d_in[27];
    const float* wpb = (const float*)d_in[28];
    const float* spb = (const float*)d_in[29];
    const float* bpb = (const float*)d_in[30];

    float* prob     = (float*)d_out;
    float* prob_nms = (float*)d_out + NHW;
    float* maskp    = (float*)d_out + 2 * NHW;

    // --- carve scratch ---
    char* base = (char*)d_ws;
    size_t off = 0;
    auto carve = [&](size_t bytes) -> void* {
        void* p = base + off;
        off += (bytes + 255) & ~(size_t)255;
        return p;
    };
    // padded act buffers: max extent = N*(482*642)*64 halves, + tail slack
    size_t actHalves = (size_t)N * 482 * 642 * 64 + (1u << 19);
    _Float16* actA = (_Float16*)carve(actHalves * 2);
    _Float16* actB = (_Float16*)carve(actHalves * 2);
    const int cin_[8]  = {64, 64, 64, 64, 128, 128, 128, 128};
    const int cout_[8] = {64, 64, 64, 128, 128, 128, 128, 256};
    _Float16* wpk[8];
    for (int i = 0; i < 8; ++i) wpk[i] = (_Float16*)carve(packElems(cin_[i], cout_[i]) * 2);
    _Float16* wpbk = (_Float16*)carve(65 * 256 * 2);
    float* cand_s  = (float*)carve((size_t)N * CAP * 4);
    int*   cand_i  = (int*)carve((size_t)N * CAP * 4);
    int*   cand_c  = (int*)carve(64);

    // --- prep ---
    const float* wsrc[8] = {w[1], w[2], w[3], w[4], w[5], w[6], w[7], wpa};
    for (int i = 0; i < 8; ++i) {
        size_t pe = packElems(cin_[i], cout_[i]);
        pack_w_kernel<<<(unsigned)((pe + 255) / 256), 256, 0, stream>>>(wsrc[i], wpk[i],
                                                                        cin_[i], cout_[i]);
    }
    pack_wpb_kernel<<<(65 * 256 + 255) / 256, 256, 0, stream>>>(wpb, wpbk);
    zero_f32_kernel<<<2048, 256, 0, stream>>>(prob_nms, 2 * NHW);
    init_cnt_kernel<<<1, 32, 0, stream>>>(cand_c, N);

    // --- backbone (padded channel-last activations) ---
    conv1_kernel<<<4800, 256, 0, stream>>>(x, w[0], sc[0], bi[0], actA, 480, 640, N);
    zero_border_kernel<<<1024, 256, 0, stream>>>(actA, 480, 640, 64, N);
    conv3x3_cl_wmma<64, 64><<<dim3(10, 480, N * 2), 256, 0, stream>>>(
        actA, wpk[0], sc[1], bi[1], actB, 480, 640, N);
    pool2_kernel<<<2048, 256, 0, stream>>>(actB, actA, 64, 480, 640, N);
    zero_border_kernel<<<512, 256, 0, stream>>>(actA, 240, 320, 64, N);
    conv3x3_cl_wmma<64, 64><<<dim3(5, 240, N * 2), 256, 0, stream>>>(
        actA, wpk[1], sc[2], bi[2], actB, 240, 320, N);
    zero_border_kernel<<<512, 256, 0, stream>>>(actB, 240, 320, 64, N);
    conv3x3_cl_wmma<64, 64><<<dim3(5, 240, N * 2), 256, 0, stream>>>(
        actB, wpk[2], sc[3], bi[3], actA, 240, 320, N);
    pool2_kernel<<<1024, 256, 0, stream>>>(actA, actB, 64, 240, 320, N);
    zero_border_kernel<<<256, 256, 0, stream>>>(actB, 120, 160, 64, N);
    conv3x3_cl_wmma<64, 128><<<dim3(3, 120, N * 4), 256, 0, stream>>>(
        actB, wpk[3], sc[4], bi[4], actA, 120, 160, N);
    zero_border_kernel<<<256, 256, 0, stream>>>(actA, 120, 160, 128, N);
    conv3x3_cl_wmma<128, 128><<<dim3(3, 120, N * 4), 256, 0, stream>>>(
        actA, wpk[4], sc[5], bi[5], actB, 120, 160, N);
    pool2_kernel<<<512, 256, 0, stream>>>(actB, actA, 128, 120, 160, N);
    zero_border_kernel<<<128, 256, 0, stream>>>(actA, 60, 80, 128, N);
    conv3x3_cl_wmma<128, 128><<<dim3(2, 60, N * 4), 256, 0, stream>>>(
        actA, wpk[5], sc[6], bi[6], actB, 60, 80, N);
    zero_border_kernel<<<128, 256, 0, stream>>>(actB, 60, 80, 128, N);
    conv3x3_cl_wmma<128, 128><<<dim3(2, 60, N * 4), 256, 0, stream>>>(
        actB, wpk[6], sc[7], bi[7], actA, 60, 80, N);
    zero_border_kernel<<<128, 256, 0, stream>>>(actA, 60, 80, 128, N);
    conv3x3_cl_wmma<128, 256><<<dim3(2, 60, N * 8), 256, 0, stream>>>(
        actA, wpk[7], spa, bpa, actB, 60, 80, N);

    // --- head + NMS ---
    head_kernel<<<(N * 60 * 80 + 255) / 256, 256, 0, stream>>>(actB, wpbk, spb, bpb, prob, N);
    gather_kernel<<<2048, 256, 0, stream>>>(prob, cand_s, cand_i, cand_c, N);
    nms_kernel<<<N, 1024, 0, stream>>>(cand_s, cand_i, cand_c, prob_nms, maskp);
}
